// Model_GNN_53077205844626
// MI455X (gfx1250) — compile-verified, matrix operand order checked by reading, not run on hardware
//
#include <hip/hip_runtime.h>
#include <hip/hip_bf16.h>

// ---------------------------------------------------------------------------
// Problem constants (from reference)
// ---------------------------------------------------------------------------
#define B_GRAPHS   8
#define N_NODES    4096
#define E_TOTAL    262144          // B * 32768 edges
#define K1P        3687
#define K2P        2950
#define K3P        2065
#define EPS_BN     1e-5f

// Vector / WMMA types (CDNA5 / gfx1250, wave32)
typedef __attribute__((ext_vector_type(16))) __bf16        v16bf;
typedef __attribute__((ext_vector_type(8)))  float         v8f;
typedef __attribute__((ext_vector_type(4)))  float         f32x4;
typedef __attribute__((ext_vector_type(4)))  unsigned int  u32x4;
typedef __attribute__((ext_vector_type(2)))  unsigned int  u32x2;

// GCC-style vector + address-space types matching the async builtin's signature
// (diagnostic: 'int __vector(4) __device__ *'  ==  v4i addrspace(1)*)
typedef __attribute__((__vector_size__(4 * sizeof(int)))) int vi4;
#define AS1 __attribute__((address_space(1)))
#define AS3 __attribute__((address_space(3)))

// CDNA5 async global->LDS path (ASYNCcnt, CDNA5 ISA),
// guarded so the file compiles on toolchains without the builtins.
#if defined(__AMDGCN__) && __has_builtin(__builtin_amdgcn_global_load_async_to_lds_b128) && __has_builtin(__builtin_amdgcn_s_wait_asynccnt)
#define HAVE_ASYNC_LDS 1
#else
#define HAVE_ASYNC_LDS 0
#endif

static __device__ __forceinline__ unsigned short f2bf_rne(float f) {
    union { float f; unsigned int u; } x; x.f = f;
    unsigned int r = x.u + 0x7FFFu + ((x.u >> 16) & 1u);   // round-to-nearest-even
    return (unsigned short)(r >> 16);
}

static __device__ __forceinline__ float sigmoidf(float v) {
    return 1.0f / (1.0f + __expf(-v));
}

// copy one 16-byte bf16 chunk global -> LDS
static __device__ __forceinline__ void stage_chunk(const unsigned short* __restrict__ g,
                                                   unsigned short* __restrict__ l)
{
#if HAVE_ASYNC_LDS
    __builtin_amdgcn_global_load_async_to_lds_b128(
        (AS1 vi4*)(unsigned long long)(const void*)g,
        (AS3 vi4*)(unsigned)(unsigned long long)(void*)l,
        0, 0);
#else
    *reinterpret_cast<u32x4*>(l) = *reinterpret_cast<const u32x4*>(g);
#endif
}

static __device__ __forceinline__ void stage_wait()
{
#if HAVE_ASYNC_LDS
    __builtin_amdgcn_s_wait_asynccnt(0);
#endif
    __syncthreads();
}

// ---------------------------------------------------------------------------
// WMMA GEMM (bf16 in, f32 out): C[M,N] = act( A[M,K] @ W[K,N] + bias[N] )
//   A   : bf16 [Mpad x K] row-major (activations, pre-converted)
//   Wt  : bf16 [N x K]    (weights, pre-transposed during conversion)
// PRECONDITIONS: N%64==0, K%32==0, buffers padded to 64-row tiles -> no guards.
// 8 wave32 / block, 64x64 C tile, 2 v_wmma per wave per K-step, double-buffered
// LDS with async global->LDS staging (1 b128 chunk per thread per tile).
// ---------------------------------------------------------------------------
#define TM 64
#define TN 64
#define TK 32
#define LP 40     // LDS pitch in ushort: 80-byte rows, 16B aligned

__global__ __launch_bounds__(256) void gemm_bf16(
    const unsigned short* __restrict__ A, const unsigned short* __restrict__ Wt,
    const float* __restrict__ bias, float* __restrict__ C,
    int M, int K, int N, int act)
{
    __shared__ __align__(16) unsigned short As[2][TM * LP];
    __shared__ __align__(16) unsigned short Bs[2][TN * LP];

    const int tid  = threadIdx.x;
    const int wave = tid >> 5;
    const int lane = tid & 31;
    const int wm   = wave >> 2;          // 0..1
    const int wn   = wave & 3;           // 0..3
    const int mBase = blockIdx.y * TM;
    const int nBase = blockIdx.x * TN;

    // staging: thread -> one 8-ushort chunk of the 64x32 tile
    const int sr = tid >> 2;             // row 0..63
    const int sc = (tid & 3) << 3;       // k-col 0,8,16,24
    const unsigned short* gA = &A [(size_t)(mBase + sr) * K + sc];
    const unsigned short* gB = &Wt[(size_t)(nBase + sr) * K + sc];
    unsigned short* lA0 = &As[0][sr * LP + sc];
    unsigned short* lB0 = &Bs[0][sr * LP + sc];
    unsigned short* lA1 = &As[1][sr * LP + sc];
    unsigned short* lB1 = &Bs[1][sr * LP + sc];

    v8f acc0 = {}, acc1 = {};

    stage_chunk(gA, lA0);
    stage_chunk(gB, lB0);
    stage_wait();

    // fragment addressing (ISA 7.12.2): lanes 0-15 K={0..7,16..23}, 16-31 K={8..15,24..31}
    const int fr    = lane & 15;
    const int kbase = (lane < 16) ? 0 : 8;

    int buf = 0;
    for (int kk = 0; kk < K; kk += TK) {
        const int nk = kk + TK;
        if (nk < K) {                     // prefetch next tile into other buffer
            stage_chunk(gA + nk, buf ? lA0 : lA1);
            stage_chunk(gB + nk, buf ? lB0 : lB1);
        }

        union { v16bf v; u32x4 q[2]; } af0, af1, bf;
        const unsigned short* bp = &Bs[buf][(wn * 16 + fr) * LP];
        bf.q[0]  = *reinterpret_cast<const u32x4*>(bp + kbase);
        bf.q[1]  = *reinterpret_cast<const u32x4*>(bp + 16 + kbase);
        const unsigned short* a0 = &As[buf][(wm * 16 + fr) * LP];
        af0.q[0] = *reinterpret_cast<const u32x4*>(a0 + kbase);
        af0.q[1] = *reinterpret_cast<const u32x4*>(a0 + 16 + kbase);
        const unsigned short* a1 = a0 + 32 * LP;
        af1.q[0] = *reinterpret_cast<const u32x4*>(a1 + kbase);
        af1.q[1] = *reinterpret_cast<const u32x4*>(a1 + 16 + kbase);

        acc0 = __builtin_amdgcn_wmma_f32_16x16x32_bf16(
                   false, af0.v, false, bf.v, (short)0, acc0, false, false);
        acc1 = __builtin_amdgcn_wmma_f32_16x16x32_bf16(
                   false, af1.v, false, bf.v, (short)0, acc1, false, false);

        stage_wait();                     // next tile resident + all waves done reading
        buf ^= 1;
    }

    // epilogue: C layout: VGPR r of lane L -> M = r + 8*(L>=16), N = L&15
    const int cn = nBase + wn * 16 + (lane & 15);
    const float bv = bias ? bias[cn] : 0.0f;
    const int rOff = (lane >> 4) << 3;
    #pragma unroll
    for (int r = 0; r < 8; ++r) {
        int cm = mBase + wm * 16 + r + rOff;
        float v0 = acc0[r] + bv;
        float v1 = acc1[r] + bv;
        if (act == 1)      { v0 = fmaxf(v0, 0.0f);  v1 = fmaxf(v1, 0.0f); }
        else if (act == 2) { v0 = sigmoidf(v0);     v1 = sigmoidf(v1); }
        C[(size_t)cm * N + cn]        = v0;
        C[(size_t)(cm + 32) * N + cn] = v1;
    }
}

// f32 -> packed bf16 (4 elems/thread)
__global__ void cvt_bf16(const float* __restrict__ in, unsigned short* __restrict__ out,
                         int n4)
{
    int i = blockIdx.x * blockDim.x + threadIdx.x;
    if (i >= n4) return;
    f32x4 v = *reinterpret_cast<const f32x4*>(&in[i << 2]);
    u32x2 p;
    p.x = (unsigned)f2bf_rne(v[0]) | ((unsigned)f2bf_rne(v[1]) << 16);
    p.y = (unsigned)f2bf_rne(v[2]) | ((unsigned)f2bf_rne(v[3]) << 16);
    *reinterpret_cast<u32x2*>(&out[i << 2]) = p;
}

// W[K,N] f32 -> Wt[N,K] bf16 (transpose during conversion; K is a power of two)
__global__ void cvt_bf16_t(const float* __restrict__ W, unsigned short* __restrict__ wt,
                           int kShift, int kMask, int N, int total)
{
    int i = blockIdx.x * blockDim.x + threadIdx.x;
    if (i >= total) return;
    int n = i >> kShift, k = i & kMask;
    wt[i] = f2bf_rne(W[(size_t)k * N + n]);
}

// ---------------------------------------------------------------------------
// ResGatedGraphConv edge scatter (4 ch/thread, several edges/block):
//   agg[dst] += sigmoid(k[dst] + q[src]) * v[src] * mask   (agg preloaded w/ skip(x))
// ---------------------------------------------------------------------------
__global__ __launch_bounds__(256) void rgc_edge(
    const float* __restrict__ k, const float* __restrict__ q,
    const float* __restrict__ v,
    const int* __restrict__ src, const int* __restrict__ dst,
    const float* __restrict__ mask, float* __restrict__ agg,
    int C, int tpeShift, int E)
{
    const int le = threadIdx.x >> tpeShift;
    const int c4 = (threadIdx.x & ((1 << tpeShift) - 1)) << 2;
    const int e  = blockIdx.x * (256 >> tpeShift) + le;
    if (e >= E) return;
    const float m = mask[e];
    if (m == 0.0f) return;
    const int s = src[e], d = dst[e];
    f32x4 kv = *reinterpret_cast<const f32x4*>(&k[(size_t)d * C + c4]);
    f32x4 qv = *reinterpret_cast<const f32x4*>(&q[(size_t)s * C + c4]);
    f32x4 vv = *reinterpret_cast<const f32x4*>(&v[(size_t)s * C + c4]);
    float* ap = &agg[(size_t)d * C + c4];
    #pragma unroll
    for (int j = 0; j < 4; ++j)
        atomicAdd(&ap[j], sigmoidf(kv[j] + qv[j]) * vv[j] * m);
}

// BatchNorm (eval) + ReLU, in place, f32x4
__global__ void bn_relu(float* __restrict__ x, const float* __restrict__ g,
                        const float* __restrict__ b, int cMask, int total4)
{
    int i = blockIdx.x * blockDim.x + threadIdx.x;
    if (i >= total4) return;
    const float rs = rsqrtf(1.0f + EPS_BN);
    int base = i << 2;
    int c = base & cMask;
    f32x4 v  = *reinterpret_cast<f32x4*>(&x[base]);
    f32x4 gv = *reinterpret_cast<const f32x4*>(&g[c]);
    f32x4 bv = *reinterpret_cast<const f32x4*>(&b[c]);
    #pragma unroll
    for (int j = 0; j < 4; ++j) v[j] = fmaxf(gv[j] * (v[j] * rs) + bv[j], 0.0f);
    *reinterpret_cast<f32x4*>(&x[base]) = v;
}

// ||w||_2 (single block)
__global__ void wnorm_kernel(const float* __restrict__ w, int C, float* __restrict__ out)
{
    __shared__ float red[256];
    float s = 0.0f;
    for (int i = threadIdx.x; i < C; i += 256) s += w[i] * w[i];
    red[threadIdx.x] = s;
    __syncthreads();
    for (int st = 128; st > 0; st >>= 1) {
        if (threadIdx.x < st) red[threadIdx.x] += red[threadIdx.x + st];
        __syncthreads();
    }
    if (threadIdx.x == 0) out[0] = sqrtf(red[0]);
}

// score_i = tanh( x_i . w / ||w|| )
__global__ void score_kernel(const float* __restrict__ x, const float* __restrict__ w,
                             const float* __restrict__ nrm, float* __restrict__ score,
                             int M, int C)
{
    int i = blockIdx.x * blockDim.x + threadIdx.x;
    if (i >= M) return;
    float s = 0.0f;
    for (int c = 0; c < C; c += 4) {
        f32x4 xv = *reinterpret_cast<const f32x4*>(&x[(size_t)i * C + c]);
        f32x4 wv = *reinterpret_cast<const f32x4*>(&w[c]);
        s += xv[0]*wv[0] + xv[1]*wv[1] + xv[2]*wv[2] + xv[3]*wv[3];
    }
    score[i] = tanhf(s / nrm[0]);
}

// per-graph top-k via bitonic sort of 4096 (score,idx) pairs in LDS (32 KB)
__global__ __launch_bounds__(1024) void topk_sort(
    const float* __restrict__ score, int* __restrict__ perm,
    float* __restrict__ pscore, int n_per, int k)
{
    __shared__ float sv[4096];
    __shared__ int   si[4096];
    const int g = blockIdx.x;

    for (int i = threadIdx.x; i < 4096; i += 1024) {
        if (i < n_per) { sv[i] = score[(size_t)g * n_per + i]; si[i] = i; }
        else           { sv[i] = -1e30f;                        si[i] = 0x7fffffff; }
    }
    __syncthreads();

    for (int size = 2; size <= 4096; size <<= 1) {
        for (int stride = size >> 1; stride > 0; stride >>= 1) {
            #pragma unroll 2
            for (int t = threadIdx.x; t < 2048; t += 1024) {
                int i = 2 * stride * (t / stride) + (t % stride);
                int j = i + stride;
                bool desc = ((i & size) == 0);
                float a = sv[i], b = sv[j];
                bool sw = desc ? (a < b) : (a > b);
                if (sw) {
                    sv[i] = b; sv[j] = a;
                    int ti = si[i]; si[i] = si[j]; si[j] = ti;
                }
            }
            __syncthreads();
        }
    }
    for (int i = threadIdx.x; i < k; i += 1024) {
        perm[(size_t)g * k + i]   = g * n_per + si[i];
        pscore[(size_t)g * k + i] = sv[i];
    }
}

// nx[j] = x[perm[j]] * pscore[j]   (block = C/4 threads)
__global__ void gather_scale(const float* __restrict__ x, const int* __restrict__ perm,
                             const float* __restrict__ ps, float* __restrict__ nx, int C)
{
    const int j  = blockIdx.x;
    const int c4 = threadIdx.x << 2;
    const float s = ps[j];
    f32x4 v = *reinterpret_cast<const f32x4*>(&x[(size_t)perm[j] * C + c4]);
    #pragma unroll
    for (int t = 0; t < 4; ++t) v[t] *= s;
    *reinterpret_cast<f32x4*>(&nx[(size_t)j * C + c4]) = v;
}

__global__ void fill_f32(float* __restrict__ p, float v, int n)
{
    int i = blockIdx.x * blockDim.x + threadIdx.x;
    if (i < n) p[i] = v;
}

__global__ void fill_i32(int* __restrict__ p, int v, int n)
{
    int i = blockIdx.x * blockDim.x + threadIdx.x;
    if (i < n) p[i] = v;
}

__global__ void newid_scatter(const int* __restrict__ perm, int* __restrict__ newid, int n)
{
    int i = blockIdx.x * blockDim.x + threadIdx.x;
    if (i < n) newid[perm[i]] = i;
}

__global__ void edge_remap(const int* __restrict__ src, const int* __restrict__ dst,
                           const float* __restrict__ mask, const int* __restrict__ newid,
                           int* __restrict__ nsrc, int* __restrict__ ndst,
                           float* __restrict__ nmask)
{
    int e = blockIdx.x * blockDim.x + threadIdx.x;
    if (e >= E_TOTAL) return;
    int ns = newid[src[e]];
    int nd = newid[dst[e]];
    nmask[e] = mask[e] * ((ns >= 0 && nd >= 0) ? 1.0f : 0.0f);
    nsrc[e] = ns > 0 ? ns : 0;
    ndst[e] = nd > 0 ? nd : 0;
}

// per-graph [max ; mean] over k rows -> gfeat[B, 2C]
__global__ void gpool_kernel(const float* __restrict__ x, float* __restrict__ gfeat,
                             int k, int C)
{
    const int g = blockIdx.x, c = threadIdx.x;
    float mx = -1e30f, sum = 0.0f;
    for (int i = 0; i < k; ++i) {
        float v = x[(size_t)(g * k + i) * C + c];
        mx = fmaxf(mx, v);
        sum += v;
    }
    gfeat[(size_t)g * 2 * C + c]     = mx;
    gfeat[(size_t)g * 2 * C + C + c] = sum / (float)k;
}

// tiny classifier linear (+optional BN affine, + act): one thread per (m,n)
__global__ void small_linear(const float* __restrict__ A, const float* __restrict__ W,
                             const float* __restrict__ bias,
                             const float* __restrict__ g, const float* __restrict__ b,
                             float* __restrict__ out, int Mb, int K, int N, int act)
{
    int idx = blockIdx.x * blockDim.x + threadIdx.x;
    if (idx >= Mb * N) return;
    int m = idx / N, n = idx % N;
    float s = 0.0f;
    for (int kk = 0; kk < K; ++kk) s += A[(size_t)m * K + kk] * W[(size_t)kk * N + n];
    s += bias[n];
    if (g) s = g[n] * (s * rsqrtf(1.0f + EPS_BN)) + b[n];
    if (act == 1)      s = fmaxf(s, 0.0f);
    else if (act == 2) s = sigmoidf(s);
    out[idx] = s;
}

__global__ void copy_f32(const float* __restrict__ src, float* __restrict__ dst, int n)
{
    int i = blockIdx.x * blockDim.x + threadIdx.x;
    if (i < n) dst[i] = src[i];
}

// ---------------------------------------------------------------------------
// Host orchestration
// ---------------------------------------------------------------------------
struct ConvP { const float *Wk,*bk,*Wq,*bq,*Wv,*bv,*Ws,*bs; };

extern "C" void kernel_launch(void* const* d_in, const int* in_sizes, int n_in,
                              void* d_out, int out_size, void* d_ws, size_t ws_size,
                              hipStream_t stream)
{
    (void)in_sizes; (void)n_in; (void)out_size; (void)ws_size;
    auto F = [&](int i) { return (const float*)d_in[i]; };
    auto I = [&](int i) { return (const int*)d_in[i]; };

    // ---- input map (setup_inputs dict flattening order)
    const float* x_in   = F(0);
    const float* embedW = F(1);  const float* embedB = F(2);
    auto conv = [&](int base) {
        ConvP p; p.Wk=F(base); p.bk=F(base+1); p.Wq=F(base+2); p.bq=F(base+3);
                 p.Wv=F(base+4); p.bv=F(base+5); p.Ws=F(base+6); p.bs=F(base+7);
        return p;
    };
    ConvP c1a = conv(3);   const float *bn1ag=F(11), *bn1ab=F(12);
    ConvP c1b = conv(13);  const float *bn1bg=F(21), *bn1bb=F(22);
    const float* pool1w = F(23);
    ConvP c2a = conv(24);  const float *bn2ag=F(32), *bn2ab=F(33);
    ConvP c2b = conv(34);  const float *bn2bg=F(42), *bn2bb=F(43);
    const float* pool2w = F(44);
    ConvP c3a = conv(45);  const float *bn3ag=F(53), *bn3ab=F(54);
    ConvP c3b = conv(55);  const float *bn3bg=F(63), *bn3bb=F(64);
    const float* pool3w = F(65);
    const float *cls1W=F(66), *cls1b=F(67), *cbn1g=F(68), *cbn1b=F(69);
    const float *cls2W=F(70), *cls2b=F(71), *cbn2g=F(72), *cbn2b=F(73);
    const float *cls3W=F(74), *cls3b=F(75);
    const int *esrc0 = I(76), *edst0 = I(77);

    // ---- workspace bump allocator (feature buffers padded to 64-row tiles)
    char* wp = (char*)d_ws;
    auto alloc = [&](size_t bytes) -> void* {
        void* p = (void*)wp; wp += (bytes + 255) & ~(size_t)255; return p;
    };
    const int MAXN = B_GRAPHS * N_NODES;                  // 32768
    float*          featA  = (float*)alloc((size_t)MAXN * 512 * 4);
    float*          featB  = (float*)alloc((size_t)MAXN * 512 * 4);
    float*          bufK   = (float*)alloc((size_t)MAXN * 256 * 4);
    float*          bufQ   = (float*)alloc((size_t)MAXN * 256 * 4);
    float*          bufV   = (float*)alloc((size_t)MAXN * 256 * 4);
    unsigned short* abf    = (unsigned short*)alloc((size_t)MAXN * 512 * 2);
    unsigned short* wt0    = (unsigned short*)alloc((size_t)512 * 256 * 2);
    unsigned short* wt1    = (unsigned short*)alloc((size_t)512 * 256 * 2);
    unsigned short* wt2    = (unsigned short*)alloc((size_t)512 * 256 * 2);
    unsigned short* wt3    = (unsigned short*)alloc((size_t)512 * 256 * 2);
    float* score  = (float*)alloc((size_t)MAXN * 4);
    float* pscore = (float*)alloc((size_t)MAXN * 4);
    int*   perm   = (int*)  alloc((size_t)MAXN * 4);
    int*   newid  = (int*)  alloc((size_t)MAXN * 4);
    float* nrmBuf = (float*)alloc(256);
    float* mask0  = (float*)alloc((size_t)E_TOTAL * 4);
    int*   eS1    = (int*)  alloc((size_t)E_TOTAL * 4);
    int*   eD1    = (int*)  alloc((size_t)E_TOTAL * 4);
    float* eM1    = (float*)alloc((size_t)E_TOTAL * 4);
    int*   eS2    = (int*)  alloc((size_t)E_TOTAL * 4);
    int*   eD2    = (int*)  alloc((size_t)E_TOTAL * 4);
    float* eM2    = (float*)alloc((size_t)E_TOTAL * 4);
    float* gfeat  = (float*)alloc(B_GRAPHS * 512 * 4);
    float* h1     = (float*)alloc(B_GRAPHS * 256 * 4);
    float* h2     = (float*)alloc(B_GRAPHS * 128 * 4);

    auto cvtA = [&](const float* src, int Mpad, int K) {
        int n4 = (Mpad * K) >> 2;
        cvt_bf16<<<(n4 + 255) / 256, 256, 0, stream>>>(src, abf, n4);
    };
    auto cvtW = [&](const float* W, unsigned short* wt, int K, int N) {
        int total = K * N;
        cvt_bf16_t<<<(total + 255) / 256, 256, 0, stream>>>(
            W, wt, __builtin_ctz(K), K - 1, N, total);
    };
    auto gemm = [&](const unsigned short* A, const unsigned short* Wt,
                    const float* bias, float* C, int M, int K, int N, int act) {
        dim3 g(N / TN, (M + TM - 1) / TM);       // N%64==0, K%32==0 by construction
        gemm_bf16<<<g, 256, 0, stream>>>(A, Wt, bias, C, M, K, N, act);
    };
    auto rgc = [&](const float* xin, const ConvP& p, const float* bg, const float* bb,
                   float* xout, int M, int K, int C,
                   const int* es, const int* ed, const float* em) {
        int Mpad = (M + 63) & ~63;
        cvtA(xin, Mpad, K);
        cvtW(p.Wk, wt0, K, C);  gemm(abf, wt0, p.bk, bufK, M, K, C, 0);
        cvtW(p.Wq, wt1, K, C);  gemm(abf, wt1, p.bq, bufQ, M, K, C, 0);
        cvtW(p.Wv, wt2, K, C);  gemm(abf, wt2, p.bv, bufV, M, K, C, 0);
        cvtW(p.Ws, wt3, K, C);  gemm(abf, wt3, p.bs, xout, M, K, C, 0);   // agg := skip(x)
        int tpeShift = (C == 256) ? 6 : 5;                 // C/4 threads per edge
        int epb = 256 >> tpeShift;
        rgc_edge<<<(E_TOTAL + epb - 1) / epb, 256, 0, stream>>>(
            bufK, bufQ, bufV, es, ed, em, xout, C, tpeShift, E_TOTAL);
        int total4 = (M * C) >> 2;
        bn_relu<<<(total4 + 255) / 256, 256, 0, stream>>>(xout, bg, bb, C - 1, total4);
    };
    auto pool = [&](const float* xin, const float* w, float* xout,
                    int n_per, int k, int C,
                    const int* es, const int* ed, const float* em,
                    int* ns, int* nd, float* nm, bool remap) {
        wnorm_kernel<<<1, 256, 0, stream>>>(w, C, nrmBuf);
        int M = B_GRAPHS * n_per;
        score_kernel<<<(M + 255) / 256, 256, 0, stream>>>(xin, w, nrmBuf, score, M, C);
        topk_sort<<<B_GRAPHS, 1024, 0, stream>>>(score, perm, pscore, n_per, k);
        gather_scale<<<B_GRAPHS * k, C / 4, 0, stream>>>(xin, perm, pscore, xout, C);
        if (remap) {
            fill_i32<<<(M + 255) / 256, 256, 0, stream>>>(newid, -1, M);
            newid_scatter<<<(B_GRAPHS * k + 255) / 256, 256, 0, stream>>>(perm, newid, B_GRAPHS * k);
            edge_remap<<<(E_TOTAL + 255) / 256, 256, 0, stream>>>(es, ed, em, newid, ns, nd, nm);
        }
    };

    // ================= forward =================
    fill_f32<<<(E_TOTAL + 255) / 256, 256, 0, stream>>>(mask0, 1.0f, E_TOTAL);

    // embed + ReLU : [32768,128] -> [32768,512]
    cvtA(x_in, MAXN, 128);
    cvtW(embedW, wt0, 128, 512);
    gemm(abf, wt0, embedB, featA, MAXN, 128, 512, 1);

    // block 1 (512->256, 256->256), edges gen0
    rgc(featA, c1a, bn1ag, bn1ab, featB, MAXN, 512, 256, esrc0, edst0, mask0);
    rgc(featB, c1b, bn1bg, bn1bb, featA, MAXN, 256, 256, esrc0, edst0, mask0);
    // pool1: 4096 -> K1, x: featA -> featB
    pool(featA, pool1w, featB, N_NODES, K1P, 256, esrc0, edst0, mask0, eS1, eD1, eM1, true);
    int M1 = B_GRAPHS * K1P;

    // block 2 (256->128, 128->128), edges gen1
    rgc(featB, c2a, bn2ag, bn2ab, featA, M1, 256, 128, eS1, eD1, eM1);
    rgc(featA, c2b, bn2bg, bn2bb, featB, M1, 128, 128, eS1, eD1, eM1);
    // pool2: K1 -> K2, x: featB -> featA
    pool(featB, pool2w, featA, K1P, K2P, 128, eS1, eD1, eM1, eS2, eD2, eM2, true);
    int M2 = B_GRAPHS * K2P;

    // block 3 (128->256, 256->256), edges gen2
    rgc(featA, c3a, bn3ag, bn3ab, featB, M2, 128, 256, eS2, eD2, eM2);
    rgc(featB, c3b, bn3bg, bn3bb, featA, M2, 256, 256, eS2, eD2, eM2);
    // pool3: K2 -> K3 (post-pool edges unused by reference -> skip remap)
    pool(featA, pool3w, featB, K2P, K3P, 256, nullptr, nullptr, nullptr,
         nullptr, nullptr, nullptr, false);

    // global pool -> gfeat [8,512]
    gpool_kernel<<<B_GRAPHS, 256, 0, stream>>>(featB, gfeat, K3P, 256);

    // classifier
    float* outp = (float*)d_out;                 // [out(8x10) | gfeat(8x512)]
    small_linear<<<(B_GRAPHS * 256 + 255) / 256, 256, 0, stream>>>(
        gfeat, cls1W, cls1b, cbn1g, cbn1b, h1, B_GRAPHS, 512, 256, 1);
    small_linear<<<(B_GRAPHS * 128 + 255) / 256, 256, 0, stream>>>(
        h1, cls2W, cls2b, cbn2g, cbn2b, h2, B_GRAPHS, 256, 128, 1);
    small_linear<<<(B_GRAPHS * 10 + 255) / 256, 256, 0, stream>>>(
        h2, cls3W, cls3b, nullptr, nullptr, outp, B_GRAPHS, 128, 10, 2);
    copy_f32<<<(B_GRAPHS * 512 + 255) / 256, 256, 0, stream>>>(
        gfeat, outp + B_GRAPHS * 10, B_GRAPHS * 512);
}
